// DecoderRNN_8366596292805
// MI455X (gfx1250) — matrix operand order repaired and checked by reading, not compiled, not run on hardware
//
#include <hip/hip_runtime.h>
#include <hip/hip_bf16.h>
#include <stdint.h>

// ---------------------------------------------------------------------------
// Problem constants (from reference): B=256, T=24, E=512, H=512, V=10000, A=2048
// Tseq = max(lengths) = T+1 = 25 (forced in _lengths()).
// ---------------------------------------------------------------------------
constexpr int Bb = 256;
constexpr int TT = 24;
constexpr int TS = 25;     // Tseq
constexpr int Ee = 512;
constexpr int Hh = 512;
constexpr int Vv = 10000;
constexpr int Aa = 2048;
constexpr int G4 = 2048;   // 4*H

typedef __attribute__((ext_vector_type(16))) __bf16 v16bf;
typedef __attribute__((ext_vector_type(8)))  float  v8f;

union AFrag { v16bf v; uint4 q[2]; };
union BFrag { v16bf v; uint4 q[2]; };

// ---------------------------------------------------------------------------
// Workspace layout (bytes, 256B aligned)
// ---------------------------------------------------------------------------
constexpr size_t algn(size_t x) { return (x + 255) & ~size_t(255); }

constexpr size_t SZ_XB     = algn((size_t)Bb * TS * Ee * 2);     // x bf16 [B,25,E]
constexpr size_t SZ_AWN    = algn((size_t)Aa * (Ee + Hh) * 2);   // attn_W bf16 [2048,1024]
constexpr size_t SZ_AWD    = algn((size_t)Ee * (Aa + Ee) * 2);   // attd_W bf16 [512,2560]
constexpr size_t SZ_WG     = algn((size_t)G4 * (Ee + Hh) * 2);   // [W_ih|W_hh] bf16 [2048,1024]
constexpr size_t SZ_OW     = algn((size_t)Vv * Hh * 2);          // out_W bf16 [10000,512]
constexpr size_t SZ_GBIAS  = algn((size_t)G4 * 4);               // b_ih+b_hh f32
constexpr size_t SZ_INFO   = algn((size_t)Bb * 1024 * 2);        // [x_t | h] bf16
constexpr size_t SZ_CAT2   = algn((size_t)Bb * 2560 * 2);        // [x_t | attw] bf16
constexpr size_t SZ_CELLIN = algn((size_t)Bb * 1024 * 2);        // [lstm_in | h] bf16
constexpr size_t SZ_LOGITS = algn((size_t)Bb * Aa * 4);          // attn logits f32
constexpr size_t SZ_GATES  = algn((size_t)Bb * G4 * 4);          // gates f32
constexpr size_t SZ_H32    = algn((size_t)Bb * Hh * 4);
constexpr size_t SZ_C32    = algn((size_t)Bb * Hh * 4);
constexpr size_t SZ_HS     = algn((size_t)TS * Bb * Hh * 2);     // all h_t bf16
constexpr size_t SZ_OFFS   = algn((size_t)64 * 4);               // also absorbs pipeline overreach

constexpr size_t OFF_XB     = 0;
constexpr size_t OFF_AWN    = OFF_XB + SZ_XB;
constexpr size_t OFF_AWD    = OFF_AWN + SZ_AWN;
constexpr size_t OFF_WG     = OFF_AWD + SZ_AWD;
constexpr size_t OFF_OW     = OFF_WG + SZ_WG;
constexpr size_t OFF_GBIAS  = OFF_OW + SZ_OW;
constexpr size_t OFF_INFO   = OFF_GBIAS + SZ_GBIAS;
constexpr size_t OFF_CAT2   = OFF_INFO + SZ_INFO;
constexpr size_t OFF_CELLIN = OFF_CAT2 + SZ_CAT2;
constexpr size_t OFF_LOGITS = OFF_CELLIN + SZ_CELLIN;
constexpr size_t OFF_GATES  = OFF_LOGITS + SZ_LOGITS;
constexpr size_t OFF_H32    = OFF_GATES + SZ_GATES;
constexpr size_t OFF_C32    = OFF_H32 + SZ_H32;
constexpr size_t OFF_HS     = OFF_C32 + SZ_C32;
constexpr size_t OFF_OFFS   = OFF_HS + SZ_HS;

__device__ __forceinline__ float sigmoidf_(float x) { return 1.f / (1.f + __expf(-x)); }

// ---------------------------------------------------------------------------
// Elementwise prep kernels
// ---------------------------------------------------------------------------
__global__ void __launch_bounds__(256) k_f32_to_bf16(const float* __restrict__ s,
                                                     __bf16* __restrict__ d, int n) {
  int i = blockIdx.x * 256 + threadIdx.x;
  if (i < n) d[i] = (__bf16)s[i];
}

// Wg[n][0:512] = W_ih[n][:], Wg[n][512:1024] = W_hh[n][:]
__global__ void __launch_bounds__(256) k_pack_cellW(const float* __restrict__ wih,
                                                    const float* __restrict__ whh,
                                                    __bf16* __restrict__ wg) {
  int i = blockIdx.x * 256 + threadIdx.x;          // over 2048*1024
  if (i >= G4 * 1024) return;
  int n = i >> 10, k = i & 1023;
  float v = (k < Ee) ? wih[n * Ee + k] : whh[n * Hh + (k - Ee)];
  wg[i] = (__bf16)v;
}

__global__ void __launch_bounds__(256) k_bias_sum(const float* __restrict__ a,
                                                  const float* __restrict__ b,
                                                  float* __restrict__ d, int n) {
  int i = blockIdx.x * 256 + threadIdx.x;
  if (i < n) d[i] = a[i] + b[i];
}

// x[b][0][:] = features[b]; x[b][t][:] = embed_W[captions[b][t-1]]  (bf16)
__global__ void __launch_bounds__(256) k_build_x(const float* __restrict__ feat,
                                                 const float* __restrict__ embW,
                                                 const int* __restrict__ caps,
                                                 __bf16* __restrict__ xb) {
  int i = blockIdx.x * 256 + threadIdx.x;          // over 256*25*512
  if (i >= Bb * TS * Ee) return;
  int e = i % Ee;
  int r = i / Ee;
  int t = r % TS;
  int b = r / TS;
  float v;
  if (t == 0) v = feat[b * Ee + e];
  else        v = embW[(size_t)caps[b * TT + (t - 1)] * Ee + e];
  xb[i] = (__bf16)v;
}

// zero h/c state, zero h-half of cellin, copy x0 into cellin x-half
__global__ void __launch_bounds__(256) k_init_state(const __bf16* __restrict__ xb,
                                                    __bf16* __restrict__ cellin,
                                                    float* __restrict__ h32,
                                                    float* __restrict__ c32) {
  int i = blockIdx.x * 256 + threadIdx.x;          // over 256*512
  if (i >= Bb * Hh) return;
  int b = i >> 9, n = i & 511;
  h32[i] = 0.f;  c32[i] = 0.f;
  cellin[b * 1024 + 512 + n] = (__bf16)0.f;
  cellin[b * 1024 + n] = xb[((size_t)b * TS + 0) * Ee + n];
}

// batch_sizes prefix sums -> offsets[0..25]
__global__ void k_offsets(const int* __restrict__ lengths, int* __restrict__ offs) {
  __shared__ int bs[TS];
  int t = threadIdx.x;
  if (t < TS) {
    int c = 0;
    for (int b = 0; b < Bb; ++b) c += (lengths[b] > t) ? 1 : 0;
    bs[t] = c;
  }
  __syncthreads();
  if (t == 0) {
    int acc = 0;
    for (int k = 0; k < TS; ++k) { offs[k] = acc; acc += bs[k]; }
    offs[TS] = acc;
  }
}

// copy x_t into info[:,0:512] and cat2[:,0:512]
__global__ void __launch_bounds__(256) k_copy_xt(const __bf16* __restrict__ xb, int t,
                                                 __bf16* __restrict__ info,
                                                 __bf16* __restrict__ cat2) {
  int i = blockIdx.x * 256 + threadIdx.x;          // over 256*512
  if (i >= Bb * Ee) return;
  int b = i >> 9, e = i & 511;
  __bf16 v = xb[((size_t)b * TS + t) * Ee + e];
  info[b * 1024 + e] = v;
  cat2[b * 2560 + e] = v;
}

// ---------------------------------------------------------------------------
// bf16 WMMA GEMM:  C[M,N] = A[M,K] * W[N,K]^T + bias
// Block = 256 threads = 8 waves; block tile = 16 rows x 512 cols.
// Each wave owns a 16x64 strip: 4 f32 accumulators, A fragment reused 4x.
// A chunk (16x64) staged in LDS, double buffered, software pipelined.
// Branch-free K-loop: pipeline overreach on the final iteration reads/stages
// one chunk past the tile (still-mapped workspace) that is never consumed.
// 8 WMMAs per workgroup barrier.  grid.x = N/512, grid.y = M/16.
// Requires ldw == K here (true at every call site), so prefetch overreach is
// just the next weight row.  Writes f32 (Cf) or bf16 (Cb).
// ---------------------------------------------------------------------------
__global__ void __launch_bounds__(256)
k_gemm_bf16(const __bf16* __restrict__ A, int lda,
            const __bf16* __restrict__ W, int ldw,
            const float* __restrict__ bias,
            float* __restrict__ Cf, __bf16* __restrict__ Cb, int ldc,
            int K) {
  __shared__ __align__(16) __bf16 As[2][16 * 64];
  const int tid  = threadIdx.x;
  const int wave = tid >> 5;
  const int lane = tid & 31;
  const int m0 = blockIdx.y * 16;
  const int n0 = (blockIdx.x * 8 + wave) * 64;
  const int nn = lane & 15;
  const int hi = lane >> 4;

  // A staging: 256 threads * 4 bf16 (8B) = 16x64 chunk
  const int sr = tid >> 4;          // 0..15
  const int sc = (tid & 15) * 4;    // 0,4,...,60
  const __bf16* Aload = A + (size_t)(m0 + sr) * lda + sc;

  const __bf16* Bp[4];
#pragma unroll
  for (int j = 0; j < 4; ++j)
    Bp[j] = W + (size_t)(n0 + j * 16 + nn) * ldw + hi * 16;

  v8f acc[4];
#pragma unroll
  for (int j = 0; j < 4; ++j) {
    const float bv = bias[n0 + j * 16 + nn];
#pragma unroll
    for (int r = 0; r < 8; ++r) acc[j][r] = bv;
  }

  const int kiters = K >> 6;
  *(unsigned long long*)(&As[0][sr * 64 + sc]) = *(const unsigned long long*)(Aload);
  __syncthreads();

  for (int ki = 0; ki < kiters; ++ki) {
    const int buf = ki & 1;
    // software pipeline: next A chunk to regs now, to LDS after compute
    const unsigned long long a_next =
        *(const unsigned long long*)(Aload + (size_t)(ki + 1) * 64);
#pragma unroll
    for (int j = 0; j < 4; ++j)
      __builtin_prefetch(Bp[j] + (size_t)(ki + 1) * 64, 0, 1);

#pragma unroll
    for (int s = 0; s < 2; ++s) {
      const int kb = s * 32;
      AFrag af;
      af.q[0] = *(const uint4*)(&As[buf][(lane & 15) * 64 + kb + hi * 16]);
      af.q[1] = *(const uint4*)(&As[buf][(lane & 15) * 64 + kb + hi * 16 + 8]);
      BFrag bfg[4];
#pragma unroll
      for (int j = 0; j < 4; ++j)
        bfg[j].v = *(const v16bf*)(Bp[j] + (size_t)ki * 64 + kb);
#pragma unroll
      for (int j = 0; j < 4; ++j)
        acc[j] = __builtin_amdgcn_wmma_f32_16x16x32_bf16(
            false, af.v, false, bfg[j].v, (short)0, acc[j], false, false);
    }
    *(unsigned long long*)(&As[buf ^ 1][sr * 64 + sc]) = a_next;
    __syncthreads();
  }

  if (Cf) {
#pragma unroll
    for (int j = 0; j < 4; ++j) {
      const int col = n0 + j * 16 + nn;
#pragma unroll
      for (int r = 0; r < 8; ++r)
        Cf[(size_t)(m0 + r + hi * 8) * ldc + col] = acc[j][r];
    }
  } else {
#pragma unroll
    for (int j = 0; j < 4; ++j) {
      const int col = n0 + j * 16 + nn;
#pragma unroll
      for (int r = 0; r < 8; ++r)
        Cb[(size_t)(m0 + r + hi * 8) * ldc + col] = (__bf16)acc[j][r];
    }
  }
}

// ---------------------------------------------------------------------------
// Row softmax over 2048 logits, scaled by cnn_features, bf16 out into cat2[:,512:]
// one block per batch row
// ---------------------------------------------------------------------------
__global__ void __launch_bounds__(256)
k_softmax_attn(const float* __restrict__ logits, const float* __restrict__ cnn,
               __bf16* __restrict__ cat2) {
  __shared__ float red[256];
  const int b = blockIdx.x, tid = threadIdx.x;
  const float* row = logits + (size_t)b * Aa;
  float m = -1e30f;
  for (int j = tid; j < Aa; j += 256) m = fmaxf(m, row[j]);
  red[tid] = m; __syncthreads();
  for (int s = 128; s > 0; s >>= 1) {
    if (tid < s) red[tid] = fmaxf(red[tid], red[tid + s]);
    __syncthreads();
  }
  m = red[0]; __syncthreads();
  float sum = 0.f;
  for (int j = tid; j < Aa; j += 256) sum += __expf(row[j] - m);
  red[tid] = sum; __syncthreads();
  for (int s = 128; s > 0; s >>= 1) {
    if (tid < s) red[tid] += red[tid + s];
    __syncthreads();
  }
  const float inv = 1.f / red[0];
  for (int j = tid; j < Aa; j += 256) {
    float aw = __expf(row[j] - m) * inv;
    cat2[(size_t)b * 2560 + 512 + j] = (__bf16)(cnn[(size_t)b * Aa + j] * aw);
  }
}

// ---------------------------------------------------------------------------
// LSTM pointwise: gate nonlinearities + state update + length masking.
// Writes h (bf16) into info[:,512:], cellin[:,512:], and hs[t].
// ---------------------------------------------------------------------------
__global__ void __launch_bounds__(256)
k_lstm_point(const float* __restrict__ gates,
             float* __restrict__ h32, float* __restrict__ c32,
             const int* __restrict__ lengths, int t,
             __bf16* __restrict__ info, __bf16* __restrict__ cellin,
             __bf16* __restrict__ hs) {
  int i = blockIdx.x * 256 + threadIdx.x;          // over 256*512
  if (i >= Bb * Hh) return;
  int b = i >> 9, n = i & 511;
  const float* g = gates + (size_t)b * G4;
  float ig = sigmoidf_(g[n]);
  float fg = sigmoidf_(g[Hh + n]);
  float gg = tanhf(g[2 * Hh + n]);
  float og = sigmoidf_(g[3 * Hh + n]);
  float cp = c32[i], hp = h32[i];
  float c2 = fg * cp + ig * gg;
  float h2 = og * tanhf(c2);
  bool act = lengths[b] > t;
  float hn = act ? h2 : hp;
  float cn = act ? c2 : cp;
  h32[i] = hn;  c32[i] = cn;
  __bf16 hb = (__bf16)hn;
  info[b * 1024 + 512 + n]   = hb;
  cellin[b * 1024 + 512 + n] = hb;
  hs[(size_t)t * Bb * Hh + i] = hb;
}

// ---------------------------------------------------------------------------
// Packed vocab projection:  out[offs[t]+b, :] = hs[t][b] @ out_W^T + out_b
// Same 16x64-per-wave scheme; grid: (ceil(625/32), 16, 25).
// Rows b = btile*16..+15, stored only while b < batch_size[t].
// 625 N-tiles of 16: per-subtile validity masks.
// ---------------------------------------------------------------------------
__global__ void __launch_bounds__(256)
k_proj(const __bf16* __restrict__ hs, const __bf16* __restrict__ W,
       const float* __restrict__ bias, const int* __restrict__ offs,
       float* __restrict__ out) {
  __shared__ __align__(16) __bf16 As[2][16 * 64];
  const int tid  = threadIdx.x;
  const int wave = tid >> 5;
  const int lane = tid & 31;
  const int t     = blockIdx.z;
  const int btile = blockIdx.y;
  const int tile0 = (blockIdx.x * 8 + wave) * 4;   // first 16-col tile of this wave
  const int nn = lane & 15;
  const int hi = lane >> 4;
  constexpr int NT = Vv / 16;                      // 625

  const __bf16* A = hs + ((size_t)t * Bb + btile * 16) * Hh;
  const int sr = tid >> 4;
  const int sc = (tid & 15) * 4;
  const __bf16* Aload = A + (size_t)sr * Hh + sc;

  bool vld[4];
  const __bf16* Bp[4];
  v8f acc[4];
#pragma unroll
  for (int j = 0; j < 4; ++j) {
    const int nt = tile0 + j;
    vld[j] = nt < NT;
    Bp[j] = vld[j] ? (W + (size_t)(nt * 16 + nn) * Hh + hi * 16) : W;
    const float bv = vld[j] ? bias[nt * 16 + nn] : 0.f;
#pragma unroll
    for (int r = 0; r < 8; ++r) acc[j][r] = bv;
  }

  *(unsigned long long*)(&As[0][sr * 64 + sc]) = *(const unsigned long long*)(Aload);
  __syncthreads();

  const int kiters = Hh >> 6;   // 8
  for (int ki = 0; ki < kiters; ++ki) {
    const int buf = ki & 1;
    const unsigned long long a_next =
        *(const unsigned long long*)(Aload + (size_t)(ki + 1) * 64);
#pragma unroll
    for (int j = 0; j < 4; ++j)
      __builtin_prefetch(Bp[j] + (size_t)(ki + 1) * 64, 0, 1);

#pragma unroll
    for (int s = 0; s < 2; ++s) {
      const int kb = s * 32;
      AFrag af;
      af.q[0] = *(const uint4*)(&As[buf][(lane & 15) * 64 + kb + hi * 16]);
      af.q[1] = *(const uint4*)(&As[buf][(lane & 15) * 64 + kb + hi * 16 + 8]);
      BFrag bfg[4];
#pragma unroll
      for (int j = 0; j < 4; ++j)
        bfg[j].v = *(const v16bf*)(Bp[j] + (size_t)ki * 64 + kb);
#pragma unroll
      for (int j = 0; j < 4; ++j)
        acc[j] = __builtin_amdgcn_wmma_f32_16x16x32_bf16(
            false, af.v, false, bfg[j].v, (short)0, acc[j], false, false);
    }
    *(unsigned long long*)(&As[buf ^ 1][sr * 64 + sc]) = a_next;
    __syncthreads();
  }

  const int bs   = offs[t + 1] - offs[t];
  const int base = offs[t];
#pragma unroll
  for (int j = 0; j < 4; ++j) {
    if (!vld[j]) continue;
    const int col = (tile0 + j) * 16 + nn;
#pragma unroll
    for (int r = 0; r < 8; ++r) {
      const int b = btile * 16 + r + hi * 8;
      if (b < bs)
        out[(size_t)(base + b) * Vv + col] = acc[j][r];
    }
  }
}

// ---------------------------------------------------------------------------
// Host launch
// ---------------------------------------------------------------------------
static inline dim3 g1(int n) { return dim3((n + 255) / 256); }

extern "C" void kernel_launch(void* const* d_in, const int* in_sizes, int n_in,
                              void* d_out, int out_size, void* d_ws, size_t ws_size,
                              hipStream_t stream) {
  const float* features = (const float*)d_in[0];
  const float* cnnf     = (const float*)d_in[1];
  const int*   captions = (const int*)d_in[2];
  const int*   lengths  = (const int*)d_in[3];
  const float* embed_W  = (const float*)d_in[4];
  const float* W_ih     = (const float*)d_in[5];
  const float* W_hh     = (const float*)d_in[6];
  const float* b_ih     = (const float*)d_in[7];
  const float* b_hh     = (const float*)d_in[8];
  const float* attn_W   = (const float*)d_in[9];
  const float* attn_b   = (const float*)d_in[10];
  const float* attd_W   = (const float*)d_in[11];
  const float* attd_b   = (const float*)d_in[12];
  const float* out_W    = (const float*)d_in[13];
  const float* out_b    = (const float*)d_in[14];

  char* ws = (char*)d_ws;
  __bf16* xb     = (__bf16*)(ws + OFF_XB);
  __bf16* awn    = (__bf16*)(ws + OFF_AWN);
  __bf16* awd    = (__bf16*)(ws + OFF_AWD);
  __bf16* wg     = (__bf16*)(ws + OFF_WG);
  __bf16* ow     = (__bf16*)(ws + OFF_OW);
  float*  gbias  = (float*) (ws + OFF_GBIAS);
  __bf16* info   = (__bf16*)(ws + OFF_INFO);
  __bf16* cat2   = (__bf16*)(ws + OFF_CAT2);
  __bf16* cellin = (__bf16*)(ws + OFF_CELLIN);
  float*  logits = (float*) (ws + OFF_LOGITS);
  float*  gates  = (float*) (ws + OFF_GATES);
  float*  h32    = (float*) (ws + OFF_H32);
  float*  c32    = (float*) (ws + OFF_C32);
  __bf16* hs     = (__bf16*)(ws + OFF_HS);
  int*    offs   = (int*)   (ws + OFF_OFFS);
  float*  out    = (float*)d_out;

  // ---- prep ----
  k_f32_to_bf16<<<g1(Aa * (Ee + Hh)), 256, 0, stream>>>(attn_W, awn, Aa * (Ee + Hh));
  k_f32_to_bf16<<<g1(Ee * (Aa + Ee)), 256, 0, stream>>>(attd_W, awd, Ee * (Aa + Ee));
  k_f32_to_bf16<<<g1(Vv * Hh), 256, 0, stream>>>(out_W, ow, Vv * Hh);
  k_pack_cellW<<<g1(G4 * 1024), 256, 0, stream>>>(W_ih, W_hh, wg);
  k_bias_sum<<<g1(G4), 256, 0, stream>>>(b_ih, b_hh, gbias, G4);
  k_build_x<<<g1(Bb * TS * Ee), 256, 0, stream>>>(features, embed_W, captions, xb);
  k_init_state<<<g1(Bb * Hh), 256, 0, stream>>>(xb, cellin, h32, c32);
  k_offsets<<<1, 32, 0, stream>>>(lengths, offs);

  // ---- step t = 0 : cell(x0, 0, 0)  (h-half of cellin is zero) ----
  k_gemm_bf16<<<dim3(G4 / 512, Bb / 16), 256, 0, stream>>>(
      cellin, 1024, wg, 1024, gbias, gates, nullptr, G4, 1024);
  k_lstm_point<<<g1(Bb * Hh), 256, 0, stream>>>(gates, h32, c32, lengths, 0,
                                                info, cellin, hs);

  // ---- steps t = 1 .. 24 ----
  for (int t = 1; t < TS; ++t) {
    k_copy_xt<<<g1(Bb * Ee), 256, 0, stream>>>(xb, t, info, cat2);
    // attn logits = info @ attn_W^T + attn_b        [256,2048], K=1024
    k_gemm_bf16<<<dim3(Aa / 512, Bb / 16), 256, 0, stream>>>(
        info, 1024, awn, 1024, attn_b, logits, nullptr, Aa, 1024);
    k_softmax_attn<<<Bb, 256, 0, stream>>>(logits, cnnf, cat2);
    // lstm_in = cat2 @ attd_W^T + attd_b  -> bf16 into cellin[:,0:512], K=2560
    k_gemm_bf16<<<dim3(Ee / 512, Bb / 16), 256, 0, stream>>>(
        cat2, 2560, awd, 2560, attd_b, nullptr, cellin, 1024, 2560);
    // gates = [lstm_in|h] @ [W_ih|W_hh]^T + (b_ih+b_hh)   [256,2048], K=1024
    k_gemm_bf16<<<dim3(G4 / 512, Bb / 16), 256, 0, stream>>>(
        cellin, 1024, wg, 1024, gbias, gates, nullptr, G4, 1024);
    k_lstm_point<<<g1(Bb * Hh), 256, 0, stream>>>(gates, h32, c32, lengths, t,
                                                  info, cellin, hs);
  }

  // ---- packed vocab projection ----
  k_proj<<<dim3((Vv / 16 + 31) / 32, Bb / 16, TS), 256, 0, stream>>>(
      hs, ow, out_b, offs, out);

  (void)in_sizes; (void)n_in; (void)out_size; (void)ws_size;
}
